// MultiHeadAttention_1726576853250
// MI455X (gfx1250) — matrix operand order repaired and checked by reading, not compiled, not run on hardware
//
#include <hip/hip_runtime.h>

#define Bn 4
#define Sn 2048
#define Dn 1024
#define Hn 16
#define DKn 64
#define DVn 64
#define NEGV -1e10f
#define NIT (Sn / 32)

typedef __bf16 bf16;
typedef __attribute__((ext_vector_type(16))) __bf16 v16bf;
typedef __attribute__((ext_vector_type(8)))  __bf16 bf16x8;
typedef __attribute__((ext_vector_type(8)))  float   v8f;
typedef __attribute__((ext_vector_type(4)))  unsigned int u32x4;
typedef __attribute__((ext_vector_type(8)))  int i32x8;
typedef __attribute__((ext_vector_type(4)))  int i32x4;
typedef int v4i __attribute__((vector_size(16)));   // matches builtin param pointee

// ---- CDNA5 feature detection (fallback chain keeps compile robust) ----
#if __has_builtin(__builtin_amdgcn_global_load_async_to_lds_b128)
#define HAVE_ASYNC 1
#else
#define HAVE_ASYNC 0
#endif

#if __has_builtin(__builtin_amdgcn_tensor_load_to_lds)
#define HAVE_TDM 1
#else
#define HAVE_TDM 0
#endif

#if HAVE_ASYNC
#if __has_builtin(__builtin_amdgcn_s_wait_asynccnt)
#define WAIT_ASYNC(N) __builtin_amdgcn_s_wait_asynccnt(N)
#else
#define WAIT_ASYNC(N) asm volatile("s_wait_asynccnt %0" ::"i"(N))
#endif
#else
#define WAIT_ASYNC(N) ((void)0)
#endif

#if HAVE_TDM
#if __has_builtin(__builtin_amdgcn_s_wait_tensorcnt)
#define WAIT_TENSOR(N) __builtin_amdgcn_s_wait_tensorcnt(N)
#else
#define WAIT_TENSOR(N) asm volatile("s_wait_tensorcnt %0" ::"i"(N))
#endif
#define NSTAGE 1   /* per-thread async ops per tile (V only; K goes via TDM) */
#else
#define WAIT_TENSOR(N) ((void)0)
#define NSTAGE 2   /* per-thread async ops per tile (K + V) */
#endif

static __device__ __forceinline__ v16bf pack16(bf16x8 lo, bf16x8 hi) {
  v16bf r;
#pragma unroll
  for (int i = 0; i < 8; ++i) { r[i] = lo[i]; r[i + 8] = hi[i]; }
  return r;
}

static __device__ __forceinline__ v8f zero8() {
  v8f z;
#pragma unroll
  for (int i = 0; i < 8; ++i) z[i] = 0.0f;
  return z;
}

// 16-byte global->LDS copy: async DMA if available, else through VGPRs.
static __device__ __forceinline__ void cp16(const bf16* g, bf16* l) {
#if HAVE_ASYNC
  __builtin_amdgcn_global_load_async_to_lds_b128(
      (__attribute__((address_space(1))) v4i*)(g),
      (__attribute__((address_space(3))) v4i*)(l), 0, 0);
#else
  *(bf16x8*)l = *(const bf16x8*)g;
#endif
}

#if HAVE_TDM
// 2D tile DMA: tile_d0 x tile_d1 elements (2B each), row stride `stride0`.
static __device__ __forceinline__ void tdm_load_2d(const void* gaddr, unsigned lds_addr,
                                                   unsigned tile_d0, unsigned tile_d1,
                                                   unsigned tensor_d0, unsigned tensor_d1,
                                                   unsigned stride0) {
  unsigned long long ga = (unsigned long long)(uintptr_t)gaddr;
  u32x4 g0;
  g0[0] = 1u;                                   // count=1, user descriptor
  g0[1] = lds_addr;                             // LDS byte address
  g0[2] = (unsigned)ga;                         // global_addr[31:0]
  g0[3] = ((unsigned)(ga >> 32) & 0x01FFFFFFu) | 0x80000000u;  // addr[56:32] | type=2
  i32x8 g1;
  g1[0] = (int)(1u << 16);                      // wg_mask=0, data_size=2B
  g1[1] = (int)((tensor_d0 & 0xFFFFu) << 16);   // tensor_dim0[15:0]
  g1[2] = (int)((tensor_d0 >> 16) | ((tensor_d1 & 0xFFFFu) << 16));
  g1[3] = (int)((tensor_d1 >> 16) | ((tile_d0 & 0xFFFFu) << 16));
  g1[4] = (int)(tile_d1 & 0xFFFFu);             // tile_dim1, tile_dim2=0
  g1[5] = (int)stride0;                         // tensor_dim0_stride[31:0]
  g1[6] = 0;
  g1[7] = 0;
  i32x4 z4; z4[0] = 0; z4[1] = 0; z4[2] = 0; z4[3] = 0;
  i32x8 z8;
#pragma unroll
  for (int i = 0; i < 8; ++i) z8[i] = 0;
  __builtin_amdgcn_tensor_load_to_lds(g0, g1, z4, z4, z8, 0);
}
#endif

// ---------------- conversion kernels ----------------
__global__ __launch_bounds__(256) void cvt_f32_bf16(const float* __restrict__ in,
                                                    bf16* __restrict__ out, int n) {
  for (int i = blockIdx.x * blockDim.x + threadIdx.x; i < n; i += gridDim.x * blockDim.x)
    out[i] = (bf16)in[i];
}

__global__ __launch_bounds__(256) void cvt_mask_u8(const int* __restrict__ in,
                                                   unsigned char* __restrict__ out, int n) {
  for (int i = blockIdx.x * blockDim.x + threadIdx.x; i < n; i += gridDim.x * blockDim.x)
    out[i] = (unsigned char)(in[i] != 0);
}

// ---------------- per-head input projection ----------------
// X:[B,S,D] bf16, W:[H,64,D] bf16.
// transposedV==0: Out[B,H,S,64].  transposedV==1: Out[B,H,64,S] (for V).
__global__ __launch_bounds__(256) void proj_head(const bf16* __restrict__ X,
                                                 const bf16* __restrict__ W,
                                                 bf16* __restrict__ Out,
                                                 int transposedV) {
  const int wave = threadIdx.x >> 5;
  const int lane = threadIdx.x & 31;
  const int hl   = lane >> 4;
  const int lrow = lane & 15;
  const int gw = blockIdx.x * 8 + wave;   // 0..8191
  const int bh = gw >> 7;                 // 0..63
  const int rt = gw & 127;
  const int b = bh >> 4, h = bh & 15;

  const bf16* Xrow = X + ((size_t)(b * Sn + rt * 16 + lrow)) * Dn;
  const bf16* Wh   = W + (size_t)h * 64 * Dn;

  v8f acc[4];
#pragma unroll
  for (int t = 0; t < 4; ++t) acc[t] = zero8();

  for (int k0 = 0; k0 < Dn; k0 += 32) {
    bf16x8 alo = *(const bf16x8*)(Xrow + k0 + hl * 8);
    bf16x8 ahi = *(const bf16x8*)(Xrow + k0 + 16 + hl * 8);
    v16bf a = pack16(alo, ahi);
#pragma unroll
    for (int t = 0; t < 4; ++t) {
      const bf16* Wrow = Wh + (size_t)(t * 16 + lrow) * Dn + k0 + hl * 8;
      v16bf bb = pack16(*(const bf16x8*)Wrow, *(const bf16x8*)(Wrow + 16));
      acc[t] = __builtin_amdgcn_wmma_f32_16x16x32_bf16(false, a, false, bb,
                                                       (short)0, acc[t], false, false);
    }
  }

  if (transposedV) {
    bf16* Ohead = Out + (size_t)(b * Hn + h) * 64 * Sn;   // [64][S]
#pragma unroll
    for (int t = 0; t < 4; ++t)
#pragma unroll
      for (int j = 0; j < 8; ++j)
        Ohead[(size_t)(t * 16 + lrow) * Sn + rt * 16 + 8 * hl + j] = (bf16)acc[t][j];
  } else {
    bf16* Orow = Out + ((size_t)(b * Hn + h) * Sn + (size_t)rt * 16) * 64;
#pragma unroll
    for (int t = 0; t < 4; ++t)
#pragma unroll
      for (int j = 0; j < 8; ++j)
        Orow[(size_t)(8 * hl + j) * 64 + t * 16 + lrow] = (bf16)acc[t][j];
  }
}

// ---------------- flash attention ----------------
// Q,K:[B,H,S,64] bf16 ; Vt:[B,H,64,S] bf16 ; mask8:[B,S,S] u8 ; Outc:[B,S,H*64] bf16
__global__ __launch_bounds__(256) void flash_attn(const bf16* __restrict__ Q,
                                                  const bf16* __restrict__ K,
                                                  const bf16* __restrict__ Vt,
                                                  const unsigned char* __restrict__ mask8,
                                                  bf16* __restrict__ Outc) {
  __shared__ __align__(16) bf16 ldsK[2][32 * 64];   // K tile, row-major (double buffered)
  __shared__ __align__(16) bf16 ldsVT[2][64 * 32];  // V^T tile [vcol][key] (double buffered)
  __shared__ __align__(16) bf16 ldsP[8][16 * 32];   // per-wave P fragment transpose

  const int tid  = threadIdx.x;
  const int wave = tid >> 5;
  const int lane = tid & 31;
  const int hl   = lane >> 4;
  const int lrow = lane & 15;

  const int blk = blockIdx.x;              // 0..1023
  const int bh  = blk >> 4;                // 0..63 (shared by all 8 waves)
  const int b = bh >> 4, h = bh & 15;
  const int rt = (blk & 15) * 8 + wave;    // query row tile 0..127

  const size_t headoff = (size_t)(b * Hn + h) * Sn * 64;
  const bf16* Qh = Q + headoff;
  const bf16* Kh = K + headoff;
  const bf16* Vh = Vt + headoff;           // [64][Sn]

  // Q tile -> two A operands.
  const bf16* Qrow = Qh + (size_t)(rt * 16 + lrow) * 64;
  v16bf qa[2];
#pragma unroll
  for (int ks = 0; ks < 2; ++ks) {
    bf16x8 lo = *(const bf16x8*)(Qrow + ks * 32 + hl * 8);
    bf16x8 hi = *(const bf16x8*)(Qrow + ks * 32 + 16 + hl * 8);
    qa[ks] = pack16(lo, hi);
  }

  float m_run[8], l_run[8];
  v8f o[4];
#pragma unroll
  for (int j = 0; j < 8; ++j) { m_run[j] = -__builtin_inff(); l_run[j] = 0.0f; }
#pragma unroll
  for (int t = 0; t < 4; ++t) o[t] = zero8();

  const float LOG2E = 1.44269504088896f;
  const float scale = 0.125f;
  const int qrow0 = rt * 16 + 8 * hl;
  const unsigned char* mbase = mask8 + (size_t)b * Sn * Sn;

  // staging coordinates
  const int vrow   = tid >> 2;             // 0..63  (V^T row = v column)
  const int vchunk = (tid & 3) * 8;        // 0,8,16,24
  const int krow   = tid >> 3;             // 0..31
  const int kcol   = (tid & 7) * 8;        // 0..56

  auto stage = [&](int t0, int buf) {
    cp16(Vh + (size_t)vrow * Sn + t0 + vchunk, &ldsVT[buf][vrow * 32 + vchunk]);
#if HAVE_TDM
    if (wave == 0)
      tdm_load_2d(Kh + (size_t)t0 * 64, (unsigned)(uintptr_t)&ldsK[buf][0],
                  /*tile*/64, 32, /*tensor*/64, 32, /*stride*/64);
#else
    cp16(Kh + (size_t)(t0 + krow) * 64 + kcol, &ldsK[buf][krow * 64 + kcol]);
#endif
  };

  stage(0, 0);

  for (int it = 0; it < NIT; ++it) {
    const int buf = it & 1;
    if (it + 1 < NIT) {
      stage((it + 1) * 32, buf ^ 1);
      WAIT_ASYNC(NSTAGE);
      if (wave == 0) WAIT_TENSOR(1);
    } else {
      WAIT_ASYNC(0);
      if (wave == 0) WAIT_TENSOR(0);
    }
    __syncthreads();

    const int t0 = it * 32;
    const bf16* kbuf = &ldsK[buf][0];
    const bf16* vbuf = &ldsVT[buf][0];

    // ---- scores: 16x32 tile, K-dim 64 ----
    v8f s0 = zero8(), s1 = zero8();
#pragma unroll
    for (int ks = 0; ks < 2; ++ks) {
      const bf16* kp0 = kbuf + lrow * 64 + ks * 32 + hl * 8;
      v16bf kb0 = pack16(*(const bf16x8*)kp0, *(const bf16x8*)(kp0 + 16));
      s0 = __builtin_amdgcn_wmma_f32_16x16x32_bf16(false, qa[ks], false, kb0,
                                                   (short)0, s0, false, false);
      const bf16* kp1 = kbuf + (16 + lrow) * 64 + ks * 32 + hl * 8;
      v16bf kb1 = pack16(*(const bf16x8*)kp1, *(const bf16x8*)(kp1 + 16));
      s1 = __builtin_amdgcn_wmma_f32_16x16x32_bf16(false, qa[ks], false, kb1,
                                                   (short)0, s1, false, false);
    }

    // ---- scale + mask ----
#pragma unroll
    for (int j = 0; j < 8; ++j) {
      const unsigned char* mp = mbase + (size_t)(qrow0 + j) * Sn + t0;
      float sv0 = s0[j] * scale; if (mp[lrow] == 0)      sv0 = NEGV;
      float sv1 = s1[j] * scale; if (mp[16 + lrow] == 0) sv1 = NEGV;
      s0[j] = sv0; s1[j] = sv1;
    }

    // ---- online softmax (reductions stay inside 16-lane fragment groups) ----
    float corr[8];
#pragma unroll
    for (int j = 0; j < 8; ++j) {
      float mv = fmaxf(s0[j], s1[j]);
#pragma unroll
      for (int off = 1; off < 16; off <<= 1) mv = fmaxf(mv, __shfl_xor(mv, off));
      float mnew = fmaxf(m_run[j], mv);
      corr[j] = exp2f((m_run[j] - mnew) * LOG2E);
      float p0 = exp2f((s0[j] - mnew) * LOG2E);
      float p1 = exp2f((s1[j] - mnew) * LOG2E);
      s0[j] = p0; s1[j] = p1;
      float ps = p0 + p1;
#pragma unroll
      for (int off = 1; off < 16; off <<= 1) ps += __shfl_xor(ps, off);
      l_run[j] = l_run[j] * corr[j] + ps;
      m_run[j] = mnew;
    }
#pragma unroll
    for (int t = 0; t < 4; ++t)
#pragma unroll
      for (int j = 0; j < 8; ++j) o[t][j] *= corr[j];

    // ---- C-layout -> A-layout for P via per-wave LDS buffer ----
    bf16* pbuf = &ldsP[wave][0];
#pragma unroll
    for (int j = 0; j < 8; ++j) {
      pbuf[(8 * hl + j) * 32 + lrow]      = (bf16)s0[j];
      pbuf[(8 * hl + j) * 32 + 16 + lrow] = (bf16)s1[j];
    }
    v16bf pa;
    {
      bf16x8 lo = *(const bf16x8*)(pbuf + lrow * 32 + hl * 8);
      bf16x8 hi = *(const bf16x8*)(pbuf + lrow * 32 + 16 + hl * 8);
      pa = pack16(lo, hi);
    }

    // ---- O += P x V (V^T rows are contiguous in LDS) ----
#pragma unroll
    for (int t = 0; t < 4; ++t) {
      const bf16* vtp = vbuf + (t * 16 + lrow) * 32 + hl * 8;
      v16bf vb = pack16(*(const bf16x8*)vtp, *(const bf16x8*)(vtp + 16));
      o[t] = __builtin_amdgcn_wmma_f32_16x16x32_bf16(false, pa, false, vb,
                                                     (short)0, o[t], false, false);
    }
    __syncthreads();
  }

  // ---- epilogue: O / l, head-major concat store ----
  bf16* orow = Outc + (size_t)(b * Sn + rt * 16) * (Hn * DVn) + h * 64;
#pragma unroll
  for (int j = 0; j < 8; ++j) {
    float inv = 1.0f / l_run[j];
#pragma unroll
    for (int t = 0; t < 4; ++t)
      orow[(size_t)(8 * hl + j) * (Hn * DVn) + t * 16 + lrow] = (bf16)(o[t][j] * inv);
  }
}

// ---------------- output projection ----------------
__global__ __launch_bounds__(256) void out_proj(const bf16* __restrict__ A,
                                                const bf16* __restrict__ W,
                                                float* __restrict__ Out) {
  const int wave = threadIdx.x >> 5;
  const int lane = threadIdx.x & 31;
  const int hl   = lane >> 4;
  const int lrow = lane & 15;
  const int gw = blockIdx.x * 8 + wave;
  const int mt = gw >> 4;
  const int nb = gw & 15;

  const bf16* Arow = A + (size_t)(mt * 16 + lrow) * 1024;
  v8f acc[4];
#pragma unroll
  for (int t = 0; t < 4; ++t) acc[t] = zero8();

  for (int k0 = 0; k0 < 1024; k0 += 32) {
    bf16x8 alo = *(const bf16x8*)(Arow + k0 + hl * 8);
    bf16x8 ahi = *(const bf16x8*)(Arow + k0 + 16 + hl * 8);
    v16bf a = pack16(alo, ahi);
#pragma unroll
    for (int t = 0; t < 4; ++t) {
      const bf16* Wrow = W + (size_t)(nb * 64 + t * 16 + lrow) * 1024 + k0 + hl * 8;
      v16bf bb = pack16(*(const bf16x8*)Wrow, *(const bf16x8*)(Wrow + 16));
      acc[t] = __builtin_amdgcn_wmma_f32_16x16x32_bf16(false, a, false, bb,
                                                       (short)0, acc[t], false, false);
    }
  }

#pragma unroll
  for (int t = 0; t < 4; ++t)
#pragma unroll
    for (int j = 0; j < 8; ++j)
      Out[(size_t)(mt * 16 + 8 * hl + j) * 1024 + nb * 64 + t * 16 + lrow] = acc[t][j];
}

// ---------------- host launcher ----------------
extern "C" void kernel_launch(void* const* d_in, const int* in_sizes, int n_in,
                              void* d_out, int out_size, void* d_ws, size_t ws_size,
                              hipStream_t stream) {
  const float* q    = (const float*)d_in[0];
  const float* k    = (const float*)d_in[1];
  const float* v    = (const float*)d_in[2];
  const int*   mask = (const int*)d_in[3];
  const float* Wq   = (const float*)d_in[4];
  const float* Wk   = (const float*)d_in[5];
  const float* Wv   = (const float*)d_in[6];
  const float* Wo   = (const float*)d_in[7];
  float* out = (float*)d_out;

  const size_t NX  = (size_t)Bn * Sn * Dn;
  const size_t NW  = (size_t)Hn * DKn * Dn;
  const size_t NWO = (size_t)Dn * Hn * DVn;
  const size_t NH  = (size_t)Bn * Hn * Sn * 64;
  const size_t NM  = (size_t)Bn * Sn * Sn;

  char* cur = (char*)d_ws;
  bf16* Xq  = (bf16*)cur; cur += NX * 2;
  bf16* Xk  = (bf16*)cur; cur += NX * 2;
  bf16* Xv  = (bf16*)cur; cur += NX * 2;
  bf16* Wqb = (bf16*)cur; cur += NW * 2;
  bf16* Wkb = (bf16*)cur; cur += NW * 2;
  bf16* Wvb = (bf16*)cur; cur += NW * 2;
  bf16* Wob = (bf16*)cur; cur += NWO * 2;
  bf16* Qh  = (bf16*)cur; cur += NH * 2;
  bf16* Kh  = (bf16*)cur; cur += NH * 2;
  bf16* Vth = (bf16*)cur; cur += NH * 2;   // V stored transposed [B,H,64,S]
  bf16* Aout = (bf16*)cur; cur += NX * 2;
  unsigned char* M8 = (unsigned char*)cur; cur += NM;

  cvt_f32_bf16<<<2048, 256, 0, stream>>>(q, Xq, (int)NX);
  cvt_f32_bf16<<<2048, 256, 0, stream>>>(k, Xk, (int)NX);
  cvt_f32_bf16<<<2048, 256, 0, stream>>>(v, Xv, (int)NX);
  cvt_f32_bf16<<<512,  256, 0, stream>>>(Wq, Wqb, (int)NW);
  cvt_f32_bf16<<<512,  256, 0, stream>>>(Wk, Wkb, (int)NW);
  cvt_f32_bf16<<<512,  256, 0, stream>>>(Wv, Wvb, (int)NW);
  cvt_f32_bf16<<<512,  256, 0, stream>>>(Wo, Wob, (int)NWO);
  cvt_mask_u8 <<<2048, 256, 0, stream>>>(mask, M8, (int)NM);

  proj_head<<<1024, 256, 0, stream>>>(Xq, Wqb, Qh, 0);
  proj_head<<<1024, 256, 0, stream>>>(Xk, Wkb, Kh, 0);
  proj_head<<<1024, 256, 0, stream>>>(Xv, Wvb, Vth, 1);

  flash_attn<<<1024, 256, 0, stream>>>(Qh, Kh, Vth, M8, Aout);

  out_proj<<<1024, 256, 0, stream>>>(Aout, Wob, out);
}
